// OctFormerBlock_83932250898949
// MI455X (gfx1250) — matrix operand order repaired and checked by reading, not compiled, not run on hardware
//
#include <hip/hip_runtime.h>
#include <hip/hip_bf16.h>
#include <math.h>

// Problem constants (from reference)
#define NW_  4096
#define KW   32
#define CH   256
#define NH   8
#define HD_  32
#define ROWS (NW_ * KW)     // 131072
#define QKVN 768
#define FFN  1024
#define RPE_ROWS 153        // 3 * (2*25+1)

typedef _Float16 v16h __attribute__((ext_vector_type(16)));
typedef _Float16 v8h  __attribute__((ext_vector_type(8)));
typedef float    v8f  __attribute__((ext_vector_type(8)));

union U16h { v16h v; _Float16 e[16]; };
union U8f  { v8f  v; float    e[8];  };

// ---------------------------------------------------------------------------
// WMMA helpers (gfx1250: D = A(16x32 f16) * B(32x16 f16) + C(16x16 f32))
// ---------------------------------------------------------------------------
__device__ __forceinline__ v8f wmma_f16(v16h a, v16h b, v8f c) {
  return __builtin_amdgcn_wmma_f32_16x16x32_f16(
      /*neg_a=*/false, a, /*neg_b=*/false, b,
      /*c_mod=*/(short)0, c, /*reuse_a=*/false, /*reuse_b=*/false);
}

// A-fragment: row-major source, lane l(0..15)=row, hi=lane>>4 selects K-half.
// halfs h=0..15 <- K = k0 + 16*hi + h (contiguous).
// p must point at (row, k0 + hi*16), 16B aligned.
__device__ __forceinline__ v16h load_a16(const _Float16* p) {
  v8h lo = *(const v8h*)p;
  v8h hh = *(const v8h*)(p + 8);
  U16h u;
#pragma unroll
  for (int i = 0; i < 8; i++) { u.e[i] = lo[i]; u.e[i + 8] = hh[i]; }
  return u.v;
}

// B-fragment from transposed weights W^T[N][K]: lane l(0..15)=col n,
// halfs h=0..15 <- K = k0 + 16*hi + h (contiguous). p at (n, k0 + hi*16).
__device__ __forceinline__ v16h load_b16(const _Float16* p) {
  v8h lo = *(const v8h*)p;
  v8h hh = *(const v8h*)(p + 8);
  U16h u;
#pragma unroll
  for (int i = 0; i < 8; i++) { u.e[i] = lo[i]; u.e[i + 8] = hh[i]; }
  return u.v;
}

__device__ __forceinline__ float gelu_tanh(float x) {
  float x3 = x * x * x;
  return 0.5f * x * (1.0f + tanhf(0.7978845608028654f * (x + 0.044715f * x3)));
}

// ---------------------------------------------------------------------------
// LayerNorm (f32 in) -> f16 out.  One wave per 256-channel row, 8 rows/block.
// ---------------------------------------------------------------------------
__global__ void __launch_bounds__(256) ln_f16_kernel(
    const float* __restrict__ x, const float* __restrict__ g,
    const float* __restrict__ b, _Float16* __restrict__ out) {
  int lane = threadIdx.x & 31;
  int row  = blockIdx.x * 8 + (threadIdx.x >> 5);
  const float4* p = (const float4*)(x + (size_t)row * CH);
  float4 v0 = p[lane];
  float4 v1 = p[lane + 32];
  float s  = v0.x + v0.y + v0.z + v0.w + v1.x + v1.y + v1.z + v1.w;
  float ss = v0.x*v0.x + v0.y*v0.y + v0.z*v0.z + v0.w*v0.w +
             v1.x*v1.x + v1.y*v1.y + v1.z*v1.z + v1.w*v1.w;
#pragma unroll
  for (int o = 16; o > 0; o >>= 1) {
    s  += __shfl_xor(s,  o, 32);
    ss += __shfl_xor(ss, o, 32);
  }
  float mean = s * (1.0f / CH);
  float var  = ss * (1.0f / CH) - mean * mean;
  float rstd = rsqrtf(var + 1e-5f);
  _Float16* op = out + (size_t)row * CH;
  int c0 = lane * 4, c1 = 128 + lane * 4;
  float a0[4] = {v0.x, v0.y, v0.z, v0.w};
  float a1[4] = {v1.x, v1.y, v1.z, v1.w};
#pragma unroll
  for (int i = 0; i < 4; i++) {
    op[c0 + i] = (_Float16)((a0[i] - mean) * rstd * g[c0 + i] + b[c0 + i]);
    op[c1 + i] = (_Float16)((a1[i] - mean) * rstd * g[c1 + i] + b[c1 + i]);
  }
}

// ---------------------------------------------------------------------------
// Weight convert+transpose: W[K][N] f32 -> WT[N][K] f16
// ---------------------------------------------------------------------------
__global__ void wt_kernel(const float* __restrict__ W, _Float16* __restrict__ WT,
                          int Kd, int N) {
  int i = blockIdx.x * 256 + threadIdx.x;
  if (i >= Kd * N) return;
  int k = i / N, n = i - k * N;
  WT[(size_t)n * Kd + k] = (_Float16)W[i];
}

// ---------------------------------------------------------------------------
// WMMA GEMM: Out[M][N] = A[M][Kc] (f16, row-major) @ WT[N][Kc]^T + epilogue
// Each wave computes a 16(M) x 64(N) strip: the A fragment is loaded once per
// K-step and reused across 4 WMMAs (4 N-tiles). 8 waves per block.
// EPI 0: +bias -> f16         (qkv)
// EPI 1: +bias +resid -> f32  (proj / fc2, residual add)
// EPI 2: +bias, gelu -> f16   (fc1)
// ---------------------------------------------------------------------------
template <int EPI>
__global__ void __launch_bounds__(256) gemm_kernel(
    const _Float16* __restrict__ A, const _Float16* __restrict__ BT,
    const float* __restrict__ bias, const float* __restrict__ resid,
    float* __restrict__ outF, _Float16* __restrict__ outH,
    int M, int N, int Kc) {
  int lane = threadIdx.x & 31;
  int strip = blockIdx.x * 8 + (threadIdx.x >> 5);
  int gn = N >> 6;                       // number of 64-wide N groups
  int mt = strip / gn, ng = strip - mt * gn;
  int m0 = mt << 4, n0 = ng << 6;
  int l = lane & 15, hi = lane >> 4;

  const _Float16* pa = A + (size_t)(m0 + l) * Kc + hi * 16;
  const _Float16* pb[4];
#pragma unroll
  for (int j = 0; j < 4; j++)
    pb[j] = BT + (size_t)(n0 + j * 16 + l) * Kc + hi * 16;

  v8f acc[4] = {{}, {}, {}, {}};
#pragma unroll 2
  for (int k0 = 0; k0 < Kc; k0 += 32) {
    v16h a = load_a16(pa); pa += 32;
    v16h b0 = load_b16(pb[0]); pb[0] += 32;
    v16h b1 = load_b16(pb[1]); pb[1] += 32;
    v16h b2 = load_b16(pb[2]); pb[2] += 32;
    v16h b3 = load_b16(pb[3]); pb[3] += 32;
    acc[0] = wmma_f16(a, b0, acc[0]);
    acc[1] = wmma_f16(a, b1, acc[1]);
    acc[2] = wmma_f16(a, b2, acc[2]);
    acc[3] = wmma_f16(a, b3, acc[3]);
  }

#pragma unroll
  for (int j = 0; j < 4; j++) {
    U8f u; u.v = acc[j];
    int n = n0 + j * 16 + l;
    float bn = bias[n];
#pragma unroll
    for (int r = 0; r < 8; r++) {
      int m = m0 + r + hi * 8;
      size_t o = (size_t)m * N + n;
      float val = u.e[r] + bn;
      if (EPI == 1)      { outF[o] = val + resid[o]; }
      else if (EPI == 2) { outH[o] = (_Float16)gelu_tanh(val); }
      else               { outH[o] = (_Float16)val; }
    }
  }
}

// ---------------------------------------------------------------------------
// Fused windowed attention: one block per window, one wave per head.
// qkv layout per token row (768): [3][H][HD]. Output f16 [ROWS][CH].
// ---------------------------------------------------------------------------
__global__ void __launch_bounds__(256) attn_kernel(
    const _Float16* __restrict__ qkv, const float* __restrict__ mask,
    const int* __restrict__ rel_pos, const float* __restrict__ rpe,
    _Float16* __restrict__ outH) {
  __shared__ float s_rpe[RPE_ROWS * NH];                 // 4896 B
  __shared__ float s_mask[KW * KW];                      // 4096 B
  __shared__ float s_S[NH][KW * 33];                     // 33792 B (padded)
  __shared__ alignas(16) _Float16 s_P[NH][KW * 40];      // 20480 B (padded, 16B rows)

  int w = blockIdx.x;
  int tid = threadIdx.x;
  for (int i = tid; i < RPE_ROWS * NH; i += 256) s_rpe[i] = rpe[i];
  const float* mk = mask + (size_t)w * KW * KW;
  for (int i = tid; i < KW * KW; i += 256) s_mask[i] = mk[i];
  __syncthreads();

  int h = tid >> 5, lane = tid & 31;
  int l = lane & 15, hi = lane >> 4;
  const _Float16* base = qkv + (size_t)w * KW * QKVN + h * HD_;
  const _Float16* qb = base;
  const _Float16* kb = base + CH;
  const _Float16* vb = base + 2 * CH;

  // ---- S = (Q * scale) @ K^T : 4 WMMAs, reduction over d (=32, one shot)
  v16h aq[2], bk[2];
#pragma unroll
  for (int mt = 0; mt < 2; mt++)
    aq[mt] = load_a16(qb + (size_t)(mt * 16 + l) * QKVN + hi * 16);
#pragma unroll
  for (int nt = 0; nt < 2; nt++)
    bk[nt] = load_b16(kb + (size_t)(nt * 16 + l) * QKVN + hi * 16);

  v8f s[2][2] = {{{}, {}}, {{}, {}}};
#pragma unroll
  for (int mt = 0; mt < 2; mt++)
#pragma unroll
    for (int nt = 0; nt < 2; nt++)
      s[mt][nt] = wmma_f16(aq[mt], bk[nt], s[mt][nt]);

  // ---- scale + RPE bias + mask -> LDS
  const float scale = 0.17677669529663687f;  // 1/sqrt(32)
  const int* rp = rel_pos + (size_t)w * KW * KW * 3;
#pragma unroll
  for (int mt = 0; mt < 2; mt++) {
#pragma unroll
    for (int nt = 0; nt < 2; nt++) {
      U8f u; u.v = s[mt][nt];
      int n = nt * 16 + l;
#pragma unroll
      for (int r = 0; r < 8; r++) {
        int m = mt * 16 + r + hi * 8;
        const int* rq = rp + ((size_t)m * KW + n) * 3;
        float bsum = 0.0f;
#pragma unroll
        for (int c = 0; c < 3; c++) {
          int ix = rq[c];
          ix = (ix < -25) ? -25 : (ix > 25 ? 25 : ix);
          ix += 25 + c * 51;
          bsum += s_rpe[ix * NH + h];
        }
        s_S[h][m * 33 + n] = u.e[r] * scale + bsum + s_mask[m * KW + n];
      }
    }
  }
  __syncthreads();

  // ---- softmax: one lane per row q
  {
    float* row = &s_S[h][lane * 33];
    float mx = -3.4e38f;
#pragma unroll
    for (int j = 0; j < KW; j++) mx = fmaxf(mx, row[j]);
    float sum = 0.0f;
#pragma unroll
    for (int j = 0; j < KW; j++) { float e = __expf(row[j] - mx); row[j] = e; sum += e; }
    float inv = 1.0f / sum;
    _Float16* prow = &s_P[h][lane * 40];
#pragma unroll
    for (int j = 0; j < KW; j++) prow[j] = (_Float16)(row[j] * inv);
  }
  __syncthreads();

  // ---- O = P @ V : A from LDS (P), B gathered from V (K-dim = token)
  v16h ap[2], bv[2];
#pragma unroll
  for (int mt = 0; mt < 2; mt++)
    ap[mt] = load_a16(&s_P[h][(mt * 16 + l) * 40 + hi * 16]);
#pragma unroll
  for (int nt = 0; nt < 2; nt++) {
    int d = nt * 16 + l;
    U16h u;
#pragma unroll
    for (int j = 0; j < 16; j++) {
      int kt = j + 16 * hi;
      u.e[j] = vb[(size_t)kt * QKVN + d];
    }
    bv[nt] = u.v;
  }
  v8f o[2][2] = {{{}, {}}, {{}, {}}};
#pragma unroll
  for (int mt = 0; mt < 2; mt++)
#pragma unroll
    for (int nt = 0; nt < 2; nt++)
      o[mt][nt] = wmma_f16(ap[mt], bv[nt], o[mt][nt]);

#pragma unroll
  for (int mt = 0; mt < 2; mt++) {
#pragma unroll
    for (int nt = 0; nt < 2; nt++) {
      U8f u; u.v = o[mt][nt];
      int d = nt * 16 + l;
#pragma unroll
      for (int r = 0; r < 8; r++) {
        int m = mt * 16 + r + hi * 8;
        outH[((size_t)w * KW + m) * CH + h * HD_ + d] = (_Float16)u.e[r];
      }
    }
  }
}

// ---------------------------------------------------------------------------
// Host pipeline
// ---------------------------------------------------------------------------
extern "C" void kernel_launch(void* const* d_in, const int* in_sizes, int n_in,
                              void* d_out, int out_size, void* d_ws, size_t ws_size,
                              hipStream_t stream) {
  const float* data   = (const float*)d_in[0];
  const float* mask   = (const float*)d_in[1];
  const int*   relp   = (const int*)d_in[2];
  const float* qkv_w  = (const float*)d_in[3];
  const float* qkv_b  = (const float*)d_in[4];
  const float* proj_w = (const float*)d_in[5];
  const float* proj_b = (const float*)d_in[6];
  const float* rpe    = (const float*)d_in[7];
  const float* ln1g   = (const float*)d_in[8];
  const float* ln1b   = (const float*)d_in[9];
  const float* ln2g   = (const float*)d_in[10];
  const float* ln2b   = (const float*)d_in[11];
  const float* fc1w   = (const float*)d_in[12];
  const float* fc1b   = (const float*)d_in[13];
  const float* fc2w   = (const float*)d_in[14];
  const float* fc2b   = (const float*)d_in[15];
  float* out = (float*)d_out;

  char* ws = (char*)d_ws;
  size_t off = 0;
  auto wsalloc = [&](size_t bytes) -> char* {
    char* p = ws + off;
    off = (off + bytes + 255) & ~(size_t)255;
    return p;
  };
  _Float16* wq_t  = (_Float16*)wsalloc((size_t)QKVN * CH * 2);
  _Float16* wp_t  = (_Float16*)wsalloc((size_t)CH * CH * 2);
  _Float16* wf1_t = (_Float16*)wsalloc((size_t)FFN * CH * 2);
  _Float16* wf2_t = (_Float16*)wsalloc((size_t)CH * FFN * 2);
  _Float16* x_h   = (_Float16*)wsalloc((size_t)ROWS * CH * 2);
  _Float16* big   = (_Float16*)wsalloc((size_t)ROWS * FFN * 2);  // qkv_h then fc1 out
  _Float16* att_h = (_Float16*)wsalloc((size_t)ROWS * CH * 2);
  _Float16* qkv_h = big;
  _Float16* fc1_h = big;

  // Weight transposes (f32 -> f16, [K][N] -> [N][K])
  wt_kernel<<<(CH * QKVN + 255) / 256, 256, 0, stream>>>(qkv_w, wq_t, CH, QKVN);
  wt_kernel<<<(CH * CH + 255) / 256, 256, 0, stream>>>(proj_w, wp_t, CH, CH);
  wt_kernel<<<(CH * FFN + 255) / 256, 256, 0, stream>>>(fc1w, wf1_t, CH, FFN);
  wt_kernel<<<(FFN * CH + 255) / 256, 256, 0, stream>>>(fc2w, wf2_t, FFN, CH);

  // LN1 -> x_h (f16)
  ln_f16_kernel<<<ROWS / 8, 256, 0, stream>>>(data, ln1g, ln1b, x_h);

  // QKV GEMM: (131072 x 256) @ (256 x 768) + b -> f16
  gemm_kernel<0><<<(ROWS / 16) * (QKVN / 64) / 8, 256, 0, stream>>>(
      x_h, wq_t, qkv_b, nullptr, nullptr, qkv_h, ROWS, QKVN, CH);

  // Fused windowed attention -> att_h (f16)
  attn_kernel<<<NW_, 256, 0, stream>>>(qkv_h, mask, relp, rpe, att_h);

  // Proj GEMM + residual -> out (f32)  [out now holds data2]
  gemm_kernel<1><<<(ROWS / 16) * (CH / 64) / 8, 256, 0, stream>>>(
      att_h, wp_t, proj_b, data, out, nullptr, ROWS, CH, CH);

  // LN2 -> x_h (f16)
  ln_f16_kernel<<<ROWS / 8, 256, 0, stream>>>(out, ln2g, ln2b, x_h);

  // FC1 GEMM + GELU -> f16
  gemm_kernel<2><<<(ROWS / 16) * (FFN / 64) / 8, 256, 0, stream>>>(
      x_h, wf1_t, fc1b, nullptr, nullptr, fc1_h, ROWS, FFN, CH);

  // FC2 GEMM + residual(data2) -> out (f32, final)
  gemm_kernel<1><<<(ROWS / 16) * (CH / 64) / 8, 256, 0, stream>>>(
      fc1_h, wf2_t, fc2b, out, out, nullptr, ROWS, CH, FFN);

  (void)in_sizes; (void)n_in; (void)out_size; (void)ws_size;
}
